// GNN_67396626809444
// MI455X (gfx1250) — matrix operand order repaired and checked by reading, not compiled, not run on hardware
//
#include <hip/hip_runtime.h>
#include <hip/hip_bf16.h>
#include <stdint.h>

// ---------------- problem constants (match reference) ----------------
#define NN      20000
#define EE      320000
#define IN_DIM  128
#define HD      128
#define NH      4
#define GG      64
#define CC      10
#define HF      512           // NH*HD
#define NEG_SLOPE 0.2f
#define BN_EPS  1e-5f
#define ENC_NEG_INF 0x007FFFFFu  // monotonic encoding of -inf

// ---------------- types for WMMA ----------------
typedef __attribute__((ext_vector_type(16))) __bf16 bf16x16;
typedef __attribute__((ext_vector_type(8)))  float  floatx8;

// ---------------- helpers ----------------
__device__ __forceinline__ unsigned short f2bf(float f) {
    unsigned u = __float_as_uint(f);
    unsigned r = u + 0x7FFFu + ((u >> 16) & 1u);   // round-to-nearest-even
    return (unsigned short)(r >> 16);
}
__device__ __forceinline__ unsigned fenc(float f) {   // monotonic float->uint
    unsigned u = __float_as_uint(f);
    return (u & 0x80000000u) ? ~u : (u | 0x80000000u);
}
__device__ __forceinline__ float fdec(unsigned e) {
    unsigned u = (e & 0x80000000u) ? (e & 0x7FFFFFFFu) : ~e;
    return __uint_as_float(u);
}

__device__ __forceinline__ bf16x16 ld_b_packed(const uint32_t* p) {
    union { uint32_t u[8]; bf16x16 v; } t;
    uint4 lo = *(const uint4*)(p);
    uint4 hi = *(const uint4*)(p + 4);
    t.u[0] = lo.x; t.u[1] = lo.y; t.u[2] = lo.z; t.u[3] = lo.w;
    t.u[4] = hi.x; t.u[5] = hi.y; t.u[6] = hi.z; t.u[7] = hi.w;
    return t.v;
}
__device__ __forceinline__ bf16x16 ld_a_row(const unsigned short* row, int kbase) {
    // 16-bit A 16x32 layout: this lane needs elems [kbase..kbase+8) and [kbase+16..kbase+24)
    union { uint32_t u[8]; bf16x16 v; } t;
    uint4 lo = *(const uint4*)(row + kbase);
    uint4 hi = *(const uint4*)(row + kbase + 16);
    t.u[0] = lo.x; t.u[1] = lo.y; t.u[2] = lo.z; t.u[3] = lo.w;
    t.u[4] = hi.x; t.u[5] = hi.y; t.u[6] = hi.z; t.u[7] = hi.w;
    return t.v;
}

// ---------------- generic fill ----------------
__global__ void k_fill_u32(uint32_t* __restrict__ p, uint32_t v, long n) {
    long i  = (long)blockIdx.x * blockDim.x + threadIdx.x;
    long st = (long)gridDim.x * blockDim.x;
    for (; i < n; i += st) p[i] = v;
}

// ---------------- embedding: x = W_embed[feat] + b_embed -> bf16 ----------------
__global__ void __launch_bounds__(256)
k_embed(const int* __restrict__ feat, const float* __restrict__ Wemb,
        const float* __restrict__ bemb, unsigned short* __restrict__ xb) {
    long i  = (long)blockIdx.x * blockDim.x + threadIdx.x;
    long st = (long)gridDim.x * blockDim.x;
    const long total = (long)NN * HD;
    for (; i < total; i += st) {
        int n = (int)(i >> 7), j = (int)(i & 127);
        float v = Wemb[(size_t)feat[n] * HD + j] + bemb[j];
        xb[i] = f2bf(v);
    }
}

// ---------------- repack W (f32 row-major KxNout) into WMMA-B lane layout (bf16) ----
// Bp dword index = ((ntile*KC + kc)*32 + lane)*8 + v
// lane: col n = ntile*16 + lane%16 ; k0 = kc*32 + (lane>=16?8:0) + 2*v + (v>=4?8:0)
__global__ void __launch_bounds__(256)
k_packW(const float* __restrict__ W, uint32_t* __restrict__ Bp, int K, int Nout) {
    const int KC = K >> 5;
    long i  = (long)blockIdx.x * blockDim.x + threadIdx.x;
    long st = (long)gridDim.x * blockDim.x;
    const long total = (long)Nout * K / 2;
    for (; i < total; i += st) {
        int v     = (int)(i & 7);
        int lane  = (int)((i >> 3) & 31);
        long rest = i >> 8;                  // = ntile*KC + kc
        int kc    = (int)(rest % KC);
        int ntile = (int)(rest / KC);
        int n  = ntile * 16 + (lane & 15);
        int k0 = kc * 32 + ((lane >> 4) << 3) + (v << 1) + ((v >> 2) << 3);
        float f0 = W[(size_t)k0 * Nout + n];
        float f1 = W[(size_t)(k0 + 1) * Nout + n];
        Bp[i] = (uint32_t)f2bf(f0) | ((uint32_t)f2bf(f1) << 16);
    }
}

// ---------------- bf16 WMMA GEMM: C[M,Nout] = A[M,K] * W, f32 accumulate -----------
// one wave -> 16x64 output tile (4 n-subtiles), 8 waves per block
__global__ void __launch_bounds__(256)
k_gemm_bf16_wmma(const unsigned short* __restrict__ A, const uint32_t* __restrict__ Bp,
                 float* __restrict__ C, int M, int K, int Nout) {
    const int lane    = threadIdx.x & 31;
    const int ngroups = Nout >> 6;
    const int KC      = K >> 5;
    const long w          = (long)blockIdx.x * 8 + (threadIdx.x >> 5);
    const long totalWaves = (long)(M >> 4) * ngroups;
    if (w >= totalWaves) return;
    const int mtile = (int)(w / ngroups);
    const int ngrp  = (int)(w % ngroups);
    const int row   = mtile * 16 + (lane & 15);
    const int kpar  = (lane >> 4) << 3;                 // 0 or 8
    const unsigned short* arow = A + (size_t)row * K;

    floatx8 acc0 = {}, acc1 = {}, acc2 = {}, acc3 = {};
    const size_t bstrideT = (size_t)KC << 8;            // dwords per n-tile
    const uint32_t* b0 = Bp + (size_t)(ngrp * 4) * bstrideT + ((size_t)lane << 3);

    for (int kc = 0; kc < KC; ++kc) {
        bf16x16 av = ld_a_row(arow, kc * 32 + kpar);
        const uint32_t* bk = b0 + ((size_t)kc << 8);
        bf16x16 bv0 = ld_b_packed(bk);
        bf16x16 bv1 = ld_b_packed(bk + bstrideT);
        bf16x16 bv2 = ld_b_packed(bk + 2 * bstrideT);
        bf16x16 bv3 = ld_b_packed(bk + 3 * bstrideT);
        acc0 = __builtin_amdgcn_wmma_f32_16x16x32_bf16(false, av, false, bv0, (short)0, acc0, false, false);
        acc1 = __builtin_amdgcn_wmma_f32_16x16x32_bf16(false, av, false, bv1, (short)0, acc1, false, false);
        acc2 = __builtin_amdgcn_wmma_f32_16x16x32_bf16(false, av, false, bv2, (short)0, acc2, false, false);
        acc3 = __builtin_amdgcn_wmma_f32_16x16x32_bf16(false, av, false, bv3, (short)0, acc3, false, false);
    }
    const int rbase = mtile * 16 + ((lane >> 4) << 3);
    const int c0    = ngrp * 64 + (lane & 15);
#pragma unroll
    for (int r = 0; r < 8; ++r) {
        size_t off = (size_t)(rbase + r) * Nout;
        C[off + c0 +  0] = acc0[r];
        C[off + c0 + 16] = acc1[r];
        C[off + c0 + 32] = acc2[r];
        C[off + c0 + 48] = acc3[r];
    }
}

// ---------------- attention dots: el/er[n,h] = <h[n,h,:], al/ar[h,:]> --------------
__global__ void __launch_bounds__(256)
k_attn_dots(const float* __restrict__ h, const float* __restrict__ al,
            const float* __restrict__ ar, float* __restrict__ el,
            float* __restrict__ er) {
    const int lane = threadIdx.x & 31;
    const long w   = (long)blockIdx.x * 8 + (threadIdx.x >> 5);
    if (w >= (long)NN * NH) return;
    const int node = (int)(w >> 2), head = (int)(w & 3);
    const float4* hp = (const float4*)(h + (size_t)node * HF + head * HD);
    const float4* ap = (const float4*)(al + head * HD);
    const float4* bp = (const float4*)(ar + head * HD);
    float4 hv = hp[lane], av = ap[lane], bv = bp[lane];
    float sl = hv.x * av.x + hv.y * av.y + hv.z * av.z + hv.w * av.w;
    float sr = hv.x * bv.x + hv.y * bv.y + hv.z * bv.z + hv.w * bv.w;
#pragma unroll
    for (int off = 16; off; off >>= 1) {
        sl += __shfl_xor(sl, off, 32);
        sr += __shfl_xor(sr, off, 32);
    }
    if (lane == 0) { el[w] = sl; er[w] = sr; }
}

// ---------------- edge pass 1: score + leakyrelu + segment max --------------------
__global__ void __launch_bounds__(256)
k_edge_max(const int* __restrict__ src, const int* __restrict__ dst,
           const float* __restrict__ el, const float* __restrict__ er,
           float* __restrict__ ev, unsigned* __restrict__ emax) {
    long i  = (long)blockIdx.x * blockDim.x + threadIdx.x;
    long st = (long)gridDim.x * blockDim.x;
    const long total = (long)EE * NH;
    for (; i < total; i += st) {
        long e = i >> 2; int hd = (int)(i & 3);
        float s = el[(size_t)src[e] * NH + hd] + er[(size_t)dst[e] * NH + hd];
        s = s > 0.f ? s : NEG_SLOPE * s;
        ev[i] = s;
        atomicMax(emax + (size_t)dst[e] * NH + hd, fenc(s));
    }
}

// ---------------- edge pass 2: exp + segment sum ----------------------------------
__global__ void __launch_bounds__(256)
k_edge_exp(const int* __restrict__ dst, float* __restrict__ ev,
           const unsigned* __restrict__ emax, float* __restrict__ denom) {
    long i  = (long)blockIdx.x * blockDim.x + threadIdx.x;
    long st = (long)gridDim.x * blockDim.x;
    const long total = (long)EE * NH;
    for (; i < total; i += st) {
        long e = i >> 2; int hd = (int)(i & 3);
        float m = fdec(emax[(size_t)dst[e] * NH + hd]);
        float w = __expf(ev[i] - m);
        ev[i] = w;
        atomicAdd(denom + (size_t)dst[e] * NH + hd, w);
    }
}

// ---------------- edge pass 3: scatter messages (one wave per edge) ---------------
__global__ void __launch_bounds__(256)
k_edge_msg(const int* __restrict__ src, const int* __restrict__ dst,
           const float* __restrict__ h, const float* __restrict__ ev,
           const float* __restrict__ denom, float* __restrict__ agg) {
    const int lane = threadIdx.x & 31;
    const long w   = (long)blockIdx.x * 8 + (threadIdx.x >> 5);
    if (w >= (long)EE) return;
    const int s = src[w], d = dst[w];
    const float4* hs = (const float4*)(h + (size_t)s * HF);
#pragma unroll
    for (int head = 0; head < NH; ++head) {
        float a = ev[w * NH + head] / (denom[(size_t)d * NH + head] + 1e-16f);
        float4 hv = hs[head * 32 + lane];
        float* ap = agg + (size_t)d * HF + head * HD + lane * 4;
        atomicAdd(ap + 0, hv.x * a);
        atomicAdd(ap + 1, hv.y * a);
        atomicAdd(ap + 2, hv.z * a);
        atomicAdd(ap + 3, hv.w * a);
    }
}

// ---------------- epilogue: +bias, relu, batchnorm -> bf16 and/or f32 -------------
__global__ void __launch_bounds__(256)
k_finish(const float* __restrict__ agg, const float* __restrict__ bias,
         const float* __restrict__ g, const float* __restrict__ be,
         const float* __restrict__ mean, const float* __restrict__ var,
         unsigned short* __restrict__ outb, float* __restrict__ outf) {
    long i  = (long)blockIdx.x * blockDim.x + threadIdx.x;
    long st = (long)gridDim.x * blockDim.x;
    const long total = (long)NN * HF;
    for (; i < total; i += st) {
        int j = (int)(i & 511);
        float v = agg[i] + bias[j];
        v = fmaxf(v, 0.f);
        v = (v - mean[j]) * rsqrtf(var[j] + BN_EPS) * g[j] + be[j];
        if (outb) outb[i] = f2bf(v);
        if (outf) outf[i] = v;
    }
}

// ---------------- graph pooling ---------------------------------------------------
__global__ void k_count(const int* __restrict__ gid, float* __restrict__ cnt) {
    int i = blockIdx.x * blockDim.x + threadIdx.x;
    if (i < NN) atomicAdd(&cnt[gid[i]], 1.0f);
}
__global__ void __launch_bounds__(256)
k_pool(const float* __restrict__ y, const int* __restrict__ gid,
       float* __restrict__ pooled) {
    long i  = (long)blockIdx.x * blockDim.x + threadIdx.x;
    long st = (long)gridDim.x * blockDim.x;
    const long total = (long)NN * HF;
    for (; i < total; i += st) {
        int node = (int)(i >> 9), j = (int)(i & 511);
        atomicAdd(&pooled[(size_t)gid[node] * HF + j], y[i]);
    }
}

// ---------------- final MLP (G=64 tiny) -------------------------------------------
__global__ void __launch_bounds__(128)
k_mlp(const float* __restrict__ pooled, const float* __restrict__ cnt,
      const float* __restrict__ Wm1, const float* __restrict__ bm1,
      const float* __restrict__ Wm2, const float* __restrict__ bm2,
      float* __restrict__ out) {
    __shared__ float hg[HF];
    __shared__ float t[HD];
    const int g = blockIdx.x;
    const float inv = 1.0f / fmaxf(cnt[g], 1.0f);
    for (int k = threadIdx.x; k < HF; k += 128) hg[k] = pooled[(size_t)g * HF + k] * inv;
    __syncthreads();
    const int j = threadIdx.x;
    float s = bm1[j];
    for (int k = 0; k < HF; ++k) s += hg[k] * Wm1[(size_t)k * HD + j];
    t[j] = fmaxf(s, 0.f);
    __syncthreads();
    if (j < CC) {
        float o = bm2[j];
        for (int k = 0; k < HD; ++k) o += t[k] * Wm2[(size_t)k * CC + j];
        out[(size_t)g * CC + j] = o;
    }
}

// ---------------- host-side orchestration -----------------------------------------
static inline long cdiv(long a, long b) { return (a + b - 1) / b; }

extern "C" void kernel_launch(void* const* d_in, const int* in_sizes, int n_in,
                              void* d_out, int out_size, void* d_ws, size_t ws_size,
                              hipStream_t stream) {
    (void)in_sizes; (void)n_in; (void)out_size; (void)ws_size;
    const int*   in_feat = (const int*)  d_in[0];
    const int*   src     = (const int*)  d_in[1];
    const int*   dst     = (const int*)  d_in[2];
    const int*   gid     = (const int*)  d_in[3];
    const float* Wemb    = (const float*)d_in[4];
    const float* bemb    = (const float*)d_in[5];
    const float* Wc1     = (const float*)d_in[6];
    const float* al1     = (const float*)d_in[7];
    const float* ar1     = (const float*)d_in[8];
    const float* bc1     = (const float*)d_in[9];
    const float* Wc2     = (const float*)d_in[10];
    const float* al2     = (const float*)d_in[11];
    const float* ar2     = (const float*)d_in[12];
    const float* bc2     = (const float*)d_in[13];
    const float* g1      = (const float*)d_in[14];
    const float* be1     = (const float*)d_in[15];
    const float* m1      = (const float*)d_in[16];
    const float* v1      = (const float*)d_in[17];
    const float* g2      = (const float*)d_in[18];
    const float* be2     = (const float*)d_in[19];
    const float* m2      = (const float*)d_in[20];
    const float* v2      = (const float*)d_in[21];
    const float* Wm1     = (const float*)d_in[22];
    const float* bm1     = (const float*)d_in[23];
    const float* Wm2     = (const float*)d_in[24];
    const float* bm2     = (const float*)d_in[25];
    float* out = (float*)d_out;

    // workspace layout
    char* W = (char*)d_ws;
    size_t o = 0;
    auto alloc = [&](size_t bytes) { size_t r = o; o += (bytes + 255) & ~(size_t)255; return r; };
    unsigned short* xb   = (unsigned short*)(W + alloc((size_t)NN * HD * 2));
    uint32_t*       pw1  = (uint32_t*)     (W + alloc((size_t)HD * HF / 2 * 4));
    uint32_t*       pw2  = (uint32_t*)     (W + alloc((size_t)HF * HF / 2 * 4));
    float*          hbuf = (float*)        (W + alloc((size_t)NN * HF * 4));
    float*          agg  = (float*)        (W + alloc((size_t)NN * HF * 4));
    unsigned short* actb = (unsigned short*)(W + alloc((size_t)NN * HF * 2));
    float*          el   = (float*)        (W + alloc((size_t)NN * NH * 4));
    float*          er   = (float*)        (W + alloc((size_t)NN * NH * 4));
    unsigned*       emax = (unsigned*)     (W + alloc((size_t)NN * NH * 4));
    float*          den  = (float*)        (W + alloc((size_t)NN * NH * 4));
    float*          ev   = (float*)        (W + alloc((size_t)EE * NH * 4));
    float*          pool = (float*)        (W + alloc((size_t)GG * HF * 4));
    float*          cnt  = (float*)        (W + alloc((size_t)GG * 4));

    const long EH = (long)EE * NH;          // 1.28M
    const long NF = (long)NN * HF;          // 10.24M
    const long gemmBlocks = cdiv((long)(NN / 16) * (HF / 64), 8);   // 1250

    // weight repack (bf16, WMMA-B lane layout)
    k_packW<<<cdiv((long)HD * HF / 2, 256), 256, 0, stream>>>(Wc1, pw1, HD, HF);
    k_packW<<<cdiv((long)HF * HF / 2, 256), 256, 0, stream>>>(Wc2, pw2, HF, HF);

    // embedding
    k_embed<<<cdiv((long)NN * HD, 256), 256, 0, stream>>>(in_feat, Wemb, bemb, xb);

    // ---------------- layer 1 ----------------
    k_fill_u32<<<cdiv((long)NN * NH, 256), 256, 0, stream>>>(emax, ENC_NEG_INF, (long)NN * NH);
    k_fill_u32<<<cdiv((long)NN * NH, 256), 256, 0, stream>>>((uint32_t*)den, 0u, (long)NN * NH);
    k_fill_u32<<<cdiv(NF, 256), 256, 0, stream>>>((uint32_t*)agg, 0u, NF);

    k_gemm_bf16_wmma<<<gemmBlocks, 256, 0, stream>>>(xb, pw1, hbuf, NN, HD, HF);
    k_attn_dots<<<cdiv((long)NN * NH, 8), 256, 0, stream>>>(hbuf, al1, ar1, el, er);
    k_edge_max<<<cdiv(EH, 256), 256, 0, stream>>>(src, dst, el, er, ev, emax);
    k_edge_exp<<<cdiv(EH, 256), 256, 0, stream>>>(dst, ev, emax, den);
    k_edge_msg<<<cdiv((long)EE, 8), 256, 0, stream>>>(src, dst, hbuf, ev, den, agg);
    k_finish<<<cdiv(NF, 256), 256, 0, stream>>>(agg, bc1, g1, be1, m1, v1, actb, nullptr);

    // ---------------- layer 2 ----------------
    k_fill_u32<<<cdiv((long)NN * NH, 256), 256, 0, stream>>>(emax, ENC_NEG_INF, (long)NN * NH);
    k_fill_u32<<<cdiv((long)NN * NH, 256), 256, 0, stream>>>((uint32_t*)den, 0u, (long)NN * NH);
    k_fill_u32<<<cdiv(NF, 256), 256, 0, stream>>>((uint32_t*)agg, 0u, NF);

    k_gemm_bf16_wmma<<<gemmBlocks, 256, 0, stream>>>(actb, pw2, hbuf, NN, HF, HF);
    k_attn_dots<<<cdiv((long)NN * NH, 8), 256, 0, stream>>>(hbuf, al2, ar2, el, er);
    k_edge_max<<<cdiv(EH, 256), 256, 0, stream>>>(src, dst, el, er, ev, emax);
    k_edge_exp<<<cdiv(EH, 256), 256, 0, stream>>>(dst, ev, emax, den);
    k_edge_msg<<<cdiv((long)EE, 8), 256, 0, stream>>>(src, dst, hbuf, ev, den, agg);
    k_finish<<<cdiv(NF, 256), 256, 0, stream>>>(agg, bc2, g2, be2, m2, v2, nullptr, agg); // in-place f32

    // ---------------- pooling + MLP ----------------
    k_fill_u32<<<cdiv((long)GG * HF, 256), 256, 0, stream>>>((uint32_t*)pool, 0u, (long)GG * HF);
    k_fill_u32<<<1, 256, 0, stream>>>((uint32_t*)cnt, 0u, GG);
    k_count<<<cdiv((long)NN, 256), 256, 0, stream>>>(gid, cnt);
    k_pool<<<cdiv(NF, 256), 256, 0, stream>>>(agg, gid, pool);
    k_mlp<<<GG, 128, 0, stream>>>(pool, cnt, Wm1, bm1, Wm2, bm2, out);
}
